// MultiHeadAttention_13572096655417
// MI455X (gfx1250) — matrix operand order repaired and checked by reading, not compiled, not run on hardware
//
#include <hip/hip_runtime.h>

typedef __attribute__((ext_vector_type(16))) _Float16 v16h;
typedef __attribute__((ext_vector_type(8)))  _Float16 v8h;
typedef __attribute__((ext_vector_type(8)))  float    v8f;

// ---------------------------------------------------------------------------
// Async global->LDS path (GLOBAL_LOAD_ASYNC_TO_LDS_B128, ASYNCcnt), guarded.
// Builtin signature (from hipcc diagnostic): (as1 int4*, as3 int4*, Ii, Ii)
// ---------------------------------------------------------------------------
#if defined(__has_builtin)
#if __has_builtin(__builtin_amdgcn_global_load_async_to_lds_b128)
#define USE_ASYNC 1
#endif
#endif
#ifndef USE_ASYNC
#define USE_ASYNC 0
#endif

#if USE_ASYNC
typedef int v4i __attribute__((vector_size(16)));
typedef __attribute__((address_space(1))) v4i* as1_v4i_p;
typedef __attribute__((address_space(3))) v4i* as3_v4i_p;

static __device__ inline void async_ld16(const _Float16* g, _Float16* l) {
  __builtin_amdgcn_global_load_async_to_lds_b128(
      (as1_v4i_p)(_Float16*)g, (as3_v4i_p)l, 0, 0);
}
#if __has_builtin(__builtin_amdgcn_s_wait_asynccnt)
#define WAIT_ASYNC(n) __builtin_amdgcn_s_wait_asynccnt(n)
#else
#define WAIT_ASYNC(n) asm volatile("s_wait_asynccnt %0" :: "i"(n) : "memory")
#endif
#endif

// ---------------------------------------------------------------------------
// WMMA operand loaders (layouts per CDNA5 ISA §7.12.2)
// ---------------------------------------------------------------------------

// A-matrix 16x32 f16, rows m0..m0+15, cols k0..k0+31, from row-major memory.
// lane<16: row=lane,    K = 0..7 (halves0-7), 16..23 (halves8-15)
// lane>=16: row=lane-16, K = 8..15, 24..31
static __device__ inline v16h load_a16x32(const _Float16* base,
                                          int ld, int m0, int k0, int lane) {
  int row = m0 + (lane & 15);
  int kb  = (lane >> 4) * 8;
  const _Float16* p = base + (size_t)row * ld + k0 + kb;
  v8h lo = *(const v8h*)(p);
  v8h hi = *(const v8h*)(p + 16);
  return __builtin_shufflevector(lo, hi, 0,1,2,3,4,5,6,7,8,9,10,11,12,13,14,15);
}

// B-matrix 32x16 f16: B[kk][n] read from memory laid out as (N,K) row-major,
// i.e. mem[n*ld + k] == B[k][n].  lane<16: N=lane, K=0..7,16..23; lane>=16: K=8..15,24..31.
// Works for both global and LDS base pointers (addrspace inferred after inline).
static __device__ inline v16h load_b32x16(const _Float16* base,
                                          int ld, int n0, int k0, int lane) {
  int n  = n0 + (lane & 15);
  int kb = (lane >> 4) * 8;
  const _Float16* p = base + (size_t)n * ld + k0 + kb;
  v8h lo = *(const v8h*)(p);
  v8h hi = *(const v8h*)(p + 16);
  return __builtin_shufflevector(lo, hi, 0,1,2,3,4,5,6,7,8,9,10,11,12,13,14,15);
}

static __device__ inline v8f wmma_f32(v16h a, v16h b, v8f c) {
  return __builtin_amdgcn_wmma_f32_16x16x32_f16(false, a, false, b, (short)0, c,
                                                false, false);
}

// ---------------------------------------------------------------------------
// fp32 -> f16 conversion
// ---------------------------------------------------------------------------
__global__ void __launch_bounds__(256) cvt_f32_f16(const float* __restrict__ in,
                                                   _Float16* __restrict__ out,
                                                   int n) {
  int i = blockIdx.x * blockDim.x + threadIdx.x;
  if (i < n) out[i] = (_Float16)in[i];
}

// ---------------------------------------------------------------------------
// Stage 1: QKV projection.  Y[m,e] = sum_k x16[m,k] * Wqkv16[e,k] + b[e]
// Scatter into Q (bh,L,64), K (bh,L,64), Vt (bh,64,L), all f16.
// Wave tile: 32 rows (m) x 64 cols (e).  All 8 waves of a block share the
// same B (weight) tile so it stays resident in the WGP cache.
// ---------------------------------------------------------------------------
__global__ void __launch_bounds__(256) qkv_gemm_kernel(
    const _Float16* __restrict__ xf, const _Float16* __restrict__ wf,
    const float* __restrict__ bias, _Float16* __restrict__ qf,
    _Float16* __restrict__ kf, _Float16* __restrict__ vtf) {
  const int lane = threadIdx.x & 31;
  const int wave = threadIdx.x >> 5;
  const int wid = blockIdx.x * 8 + wave;     // 12288 waves total
  const int mt = wid & 255;                  // 0..255  (32-row tile of x)
  const int nt = wid >> 8;                   // 0..47   (64-col tile of e)
  const int m0 = mt * 32;

  v8f acc[2][4] = {};
  for (int k0 = 0; k0 < 1024; k0 += 32) {
    v16h aA = load_a16x32(xf, 1024, m0,      k0, lane);
    v16h aB = load_a16x32(xf, 1024, m0 + 16, k0, lane);
    if (k0 + 64 < 1024) {   // prefetch next-next k-step (global_prefetch_b8)
      __builtin_prefetch(xf + (size_t)(m0 + (lane & 31)) * 1024 + k0 + 64, 0, 1);
      __builtin_prefetch(wf + (size_t)(nt * 64 + (lane & 31)) * 1024 + k0 + 64, 0, 1);
    }
#pragma unroll
    for (int s = 0; s < 4; ++s) {
      v16h b = load_b32x16(wf, 1024, nt * 64 + s * 16, k0, lane);
      acc[0][s] = wmma_f32(aA, b, acc[0][s]);
      acc[1][s] = wmma_f32(aB, b, acc[1][s]);
    }
  }

  const int n  = lane & 15;
  const int mb = (lane >> 4) * 8;
#pragma unroll
  for (int u = 0; u < 2; ++u) {
    const int mu = m0 + u * 16;
#pragma unroll
    for (int s = 0; s < 4; ++s) {
      const int e0 = nt * 64 + s * 16;   // tile lies fully in one (h, q/k/v) region
      const int h  = e0 / 192;
      const int r  = e0 % 192;
      const int t  = r / 64;             // 0=q, 1=k, 2=v
      const int d  = (r % 64) + n;
      const float bv = bias[e0 + n];
#pragma unroll
      for (int rr = 0; rr < 8; ++rr) {
        const int m  = mu + mb + rr;     // global (b,l) row
        const int bi = m >> 11;
        const int l  = m & 2047;
        const _Float16 val = (_Float16)(acc[u][s][rr] + bv);
        const size_t bh = (size_t)(bi * 16 + h);
        if (t == 0)      qf [(bh * 2048 + l) * 64 + d]  = val;
        else if (t == 1) kf [(bh * 2048 + l) * 64 + d]  = val;
        else             vtf[(bh * 64 + d) * 2048 + l]  = val;   // V transposed
      }
    }
  }
}

// ---------------------------------------------------------------------------
// Stage 2: flash attention.  Block = 4 waves = one (bh, 64-query block).
// K/V tiles are staged into LDS once per 32-key step and consumed as WMMA
// B-operands by all 4 waves.  With the async path, tile i+1 is DMA'd into
// the other LDS buffer while tile i is being consumed (ASYNCcnt pipelining).
// ---------------------------------------------------------------------------
#if USE_ASYNC
static __device__ inline void stage_async(const _Float16* kp, const _Float16* vp,
                                          _Float16* kl, _Float16* vl,
                                          int j, int tid) {
#pragma unroll
  for (int u = 0; u < 2; ++u) {
    const int c = tid + u * 128;            // 256 x 16B chunks per tile
    const int kr = c >> 3, kc = (c & 7) * 8;
    async_ld16(kp + (size_t)(j + kr) * 64 + kc, kl + c * 8);
    const int vr = c >> 2, vc = (c & 3) * 8;
    async_ld16(vp + (size_t)vr * 2048 + j + vc, vl + c * 8);
  }
}
#endif

__global__ void __launch_bounds__(128) attn_kernel(
    const _Float16* __restrict__ qf, const _Float16* __restrict__ kf,
    const _Float16* __restrict__ vtf, _Float16* __restrict__ of) {
  __shared__ __align__(32) _Float16 klds[2][32 * 64];   // K tile, rows=key
  __shared__ __align__(32) _Float16 vlds[2][64 * 32];   // Vt tile, rows=dim
  __shared__ __align__(32) _Float16 pbuf[4][16 * 32];   // per-wave P staging

  const int tid  = threadIdx.x;
  const int lane = tid & 31;
  const int wave = tid >> 5;
  const int bh   = blockIdx.x >> 5;      // 0..63
  const int qb   = blockIdx.x & 31;      // 0..31

  const _Float16* qp = qf  + (size_t)bh * 2048 * 64;
  const _Float16* kp = kf  + (size_t)bh * 2048 * 64;
  const _Float16* vp = vtf + (size_t)bh * 64 * 2048;
  const int qr0 = qb * 64 + wave * 16;

  // Q tile (16x64) as two A-operands, register-resident throughout
  const v16h a0 = load_a16x32(qp, 64, qr0, 0, lane);
  const v16h a1 = load_a16x32(qp, 64, qr0, 32, lane);

  v8f o0 = {}, o1 = {}, o2 = {}, o3 = {};
  float rowm[8], rowl[8];
#pragma unroll
  for (int r = 0; r < 8; ++r) { rowm[r] = -1e30f; rowl[r] = 0.f; }

  const float scale = 0.125f;            // 1/sqrt(64)
  _Float16* pb = pbuf[wave];
  const int n  = lane & 15;
  const int hi = lane >> 4;

#if !USE_ASYNC
  // synchronous staging coordinates (one 32B chunk x2 per thread)
  const int krow = tid >> 2, kseg = tid & 3;   // 32 rows x 4 x 32B = 4KB
  const int vrow = tid >> 1, vseg = tid & 1;   // 64 rows x 2 x 32B = 4KB
#else
  stage_async(kp, vp, klds[0], vlds[0], 0, tid);   // prologue: tile 0
#endif

  for (int jt = 0; jt < 64; ++jt) {
    const int j = jt * 32;
#if USE_ASYNC
    const int cur = jt & 1;
    const _Float16* kl = klds[cur];
    const _Float16* vl = vlds[cur];
    __syncthreads();                           // buf[cur^1] readers all done
    if (jt + 1 < 64)                           // DMA next tile while computing
      stage_async(kp, vp, klds[cur ^ 1], vlds[cur ^ 1], j + 32, tid);
    WAIT_ASYNC(4);                             // my tile-jt ops complete (in-order)
    __syncthreads();                           // all waves' pieces visible
#else
    const _Float16* kl = klds[0];
    const _Float16* vl = vlds[0];
    __syncthreads();                           // previous tile fully consumed
    *(v16h*)(klds[0] + krow * 64 + kseg * 16) =
        *(const v16h*)(kp + (size_t)(j + krow) * 64 + kseg * 16);
    *(v16h*)(vlds[0] + vrow * 32 + vseg * 16) =
        *(const v16h*)(vp + (size_t)vrow * 2048 + j + vseg * 16);
    if (j + 32 < 2048) {                       // prefetch next key tile
      __builtin_prefetch(kp + (size_t)(j + 32 + krow) * 64 + kseg * 16, 0, 1);
      __builtin_prefetch(vp + (size_t)vrow * 2048 + j + 32 + vseg * 16, 0, 1);
    }
    __syncthreads();                           // tile visible to all waves
#endif

    // S tile 16x32 (keys j..j+31) from LDS
    v8f s0 = {}, s1 = {};
    s0 = wmma_f32(a0, load_b32x16(kl, 64, 0,  0,  lane), s0);
    s0 = wmma_f32(a1, load_b32x16(kl, 64, 0,  32, lane), s0);
    s1 = wmma_f32(a0, load_b32x16(kl, 64, 16, 0,  lane), s1);
    s1 = wmma_f32(a1, load_b32x16(kl, 64, 16, 32, lane), s1);

    float alpha[8];
#pragma unroll
    for (int r = 0; r < 8; ++r) {
      float v0 = s0[r] * scale;
      float v1 = s1[r] * scale;
      float mx = fmaxf(v0, v1);
#pragma unroll
      for (int off = 1; off < 16; off <<= 1)   // row lives in one half-wave
        mx = fmaxf(mx, __shfl_xor(mx, off, 32));
      const float newm = fmaxf(rowm[r], mx);
      const float al = __expf(rowm[r] - newm);
      rowm[r] = newm;
      const float p0 = __expf(v0 - newm);
      const float p1 = __expf(v1 - newm);
      float ps = p0 + p1;
#pragma unroll
      for (int off = 1; off < 16; off <<= 1)
        ps += __shfl_xor(ps, off, 32);
      rowl[r] = rowl[r] * al + ps;
      alpha[r] = al;
      const int m = r + hi * 8;                // C-layout row
      pb[m * 32 + n]      = (_Float16)p0;
      pb[m * 32 + 16 + n] = (_Float16)p1;
    }
#pragma unroll
    for (int r = 0; r < 8; ++r) {
      o0[r] *= alpha[r]; o1[r] *= alpha[r];
      o2[r] *= alpha[r]; o3[r] *= alpha[r];
    }

    // Re-read P (16x32) from LDS in A-operand layout (same-wave DS in-order)
    const int row = lane & 15;
    const int kb8 = hi * 8;
    v8h plo = *(const v8h*)(pb + row * 32 + kb8);
    v8h phi = *(const v8h*)(pb + row * 32 + 16 + kb8);
    v16h pa = __builtin_shufflevector(plo, phi,
        0,1,2,3,4,5,6,7,8,9,10,11,12,13,14,15);

    // O += P * V from LDS:  B[kk][nn] = Vt[c*16+nn][j+kk]
    o0 = wmma_f32(pa, load_b32x16(vl, 32, 0,  0, lane), o0);
    o1 = wmma_f32(pa, load_b32x16(vl, 32, 16, 0, lane), o1);
    o2 = wmma_f32(pa, load_b32x16(vl, 32, 32, 0, lane), o2);
    o3 = wmma_f32(pa, load_b32x16(vl, 32, 48, 0, lane), o3);
  }

  // epilogue: normalize and store to (B, L, H*64) f16
  const int b = bh >> 4;
  const int h = bh & 15;
#pragma unroll
  for (int r = 0; r < 8; ++r) {
    const float inv = 1.0f / rowl[r];
    const int m = qr0 + r + hi * 8;            // query row in [0,2048)
    const size_t base = ((size_t)(b * 2048 + m)) * 1024 + h * 64 + n;
    of[base +  0] = (_Float16)(o0[r] * inv);
    of[base + 16] = (_Float16)(o1[r] * inv);
    of[base + 32] = (_Float16)(o2[r] * inv);
    of[base + 48] = (_Float16)(o3[r] * inv);
  }
}

// ---------------------------------------------------------------------------
// Stage 3: output projection.  out[m,o] = sum_h of16[m,h]*Wout16[o,h] + b[o]
// Wave tile 32x64; all 8 waves of a block share the same B (weight) tile.
// ---------------------------------------------------------------------------
__global__ void __launch_bounds__(256) out_gemm_kernel(
    const _Float16* __restrict__ of, const _Float16* __restrict__ wf,
    const float* __restrict__ bias, float* __restrict__ out) {
  const int lane = threadIdx.x & 31;
  const int wave = threadIdx.x >> 5;
  const int wid = blockIdx.x * 8 + wave;     // 4096 waves total
  const int mt = wid & 255;                  // 0..255
  const int nt = wid >> 8;                   // 0..15
  const int m0 = mt * 32;

  v8f acc[2][4] = {};
  for (int k0 = 0; k0 < 1024; k0 += 32) {
    v16h aA = load_a16x32(of, 1024, m0,      k0, lane);
    v16h aB = load_a16x32(of, 1024, m0 + 16, k0, lane);
    if (k0 + 64 < 1024) {
      __builtin_prefetch(of + (size_t)(m0 + (lane & 31)) * 1024 + k0 + 64, 0, 1);
      __builtin_prefetch(wf + (size_t)(nt * 64 + (lane & 31)) * 1024 + k0 + 64, 0, 1);
    }
#pragma unroll
    for (int s = 0; s < 4; ++s) {
      v16h b = load_b32x16(wf, 1024, nt * 64 + s * 16, k0, lane);
      acc[0][s] = wmma_f32(aA, b, acc[0][s]);
      acc[1][s] = wmma_f32(aB, b, acc[1][s]);
    }
  }

  const int n  = lane & 15;
  const int mb = (lane >> 4) * 8;
#pragma unroll
  for (int u = 0; u < 2; ++u) {
    const int mu = m0 + u * 16;
#pragma unroll
    for (int s = 0; s < 4; ++s) {
      const int n0 = nt * 64 + s * 16;
      const float bv = bias[n0 + n];
#pragma unroll
      for (int rr = 0; rr < 8; ++rr)
        out[(size_t)(mu + mb + rr) * 1024 + n0 + n] = acc[u][s][rr] + bv;
    }
  }
}

// ---------------------------------------------------------------------------
extern "C" void kernel_launch(void* const* d_in, const int* in_sizes, int n_in,
                              void* d_out, int out_size, void* d_ws, size_t ws_size,
                              hipStream_t stream) {
  const float* x    = (const float*)d_in[0];   // (4, 2048, 1024)
  const float* Wqkv = (const float*)d_in[1];   // (3072, 1024)
  const float* bqkv = (const float*)d_in[2];   // (3072)
  const float* Wout = (const float*)d_in[3];   // (1024, 1024)
  const float* bout = (const float*)d_in[4];   // (1024)
  float* out = (float*)d_out;                  // (4, 2048, 1024)

  const size_t NX = (size_t)8192 * 1024;       // 8,388,608
  const size_t NW = (size_t)3072 * 1024;       // 3,145,728
  const size_t NO = (size_t)1024 * 1024;       // 1,048,576
  const size_t NH = (size_t)64 * 2048 * 64;    // 8,388,608 per Q/K/Vt

  _Float16* xf  = (_Float16*)d_ws;             // ~88 MB total f16 workspace
  _Float16* wqf = xf  + NX;
  _Float16* wof = wqf + NW;
  _Float16* qf  = wof + NO;
  _Float16* kf  = qf  + NH;
  _Float16* vtf = kf  + NH;
  _Float16* of  = vtf + NH;

  // fp32 -> f16 conversions
  cvt_f32_f16<<<(int)((NX + 255) / 256), 256, 0, stream>>>(x,    xf,  (int)NX);
  cvt_f32_f16<<<(int)((NW + 255) / 256), 256, 0, stream>>>(Wqkv, wqf, (int)NW);
  cvt_f32_f16<<<(int)((NO + 255) / 256), 256, 0, stream>>>(Wout, wof, (int)NO);

  // QKV projection: 256 m-tiles x 48 n-tiles, 8 waves/block (B shared per block)
  qkv_gemm_kernel<<<1536, 256, 0, stream>>>(xf, wqf, bqkv, qf, kf, vtf);

  // Flash attention: 64 (b,h) * 32 query-blocks, async LDS-staged K/V
  attn_kernel<<<2048, 128, 0, stream>>>(qf, kf, vtf, of);

  // Output projection: 256 m-tiles x 16 n-tiles, 8 waves/block
  out_gemm_kernel<<<512, 256, 0, stream>>>(of, wof, bout, out);
}
